// TreeEnergyLoss_66151086293302
// MI455X (gfx1250) — compile-verified, matrix operand order checked by reading, not run on hardware
//
#include <hip/hip_runtime.h>
#include <hip/hip_bf16.h>
#include <math.h>

// ---------------------------------------------------------------------------
// Tree Energy Loss on MI455X (gfx1250, wave32).
// Pipeline per image: bilinear-resize low feats -> edge weights -> Boruvka MST
// -> BFS rooting -> level ordering -> two-pass tree filter (x2 trees) -> L1 loss.
// High-feature edge weights (256ch) use v_wmma_f32_16x16x32_f16 as a
// reduction-by-ones over channel chunks.
// Workspace demand: ~1.75 MB (see layout in kernel_launch).
// ---------------------------------------------------------------------------

typedef __attribute__((ext_vector_type(16))) _Float16 v16h;
typedef __attribute__((ext_vector_type(8)))  float    v8f;

namespace tel {
constexpr int BB   = 2;      // batch
constexpr int C    = 21;     // classes
constexpr int HH   = 64;
constexpr int WW   = 64;
constexpr int N    = HH * WW;        // 4096 nodes
constexpr int EH   = HH * (WW - 1);  // 4032 horizontal edges
constexpr int E    = 2 * EH;         // 8064 edges
constexpr int CH   = 256;    // high-feat channels
constexpr int CL   = 3;      // low-feat channels
constexpr int HL   = 256;    // low-feat spatial
constexpr float SIGMA_LOW  = 0.002f;
constexpr float SIGMA_HIGH = 1.0f;
} // namespace tel

__device__ __forceinline__ void edge_uv(int e, int& u, int& v) {
    using namespace tel;
    if (e < EH) {                    // idx[:, :-1] -> idx[:, 1:]
        int r = e / (WW - 1);
        int c = e - r * (WW - 1);
        u = r * WW + c; v = u + 1;
    } else {                         // idx[:-1, :] -> idx[1:, :]
        int t = e - EH;              // t = r*WW + c, r < HH-1
        u = t; v = t + WW;
    }
}

// ---------------------------------------------------------------------------
// Misc init
// ---------------------------------------------------------------------------
__global__ void zero2_kernel(float* a, float* b) {
    if (threadIdx.x == 0 && blockIdx.x == 0) { *a = 0.f; *b = 0.f; }
}

// ---------------------------------------------------------------------------
// Bilinear resize low_feats (B,3,256,256) -> (B,3,64,64), half-pixel centers.
// ---------------------------------------------------------------------------
__global__ void resize_low_kernel(const float* __restrict__ low, float* __restrict__ lowr) {
    using namespace tel;
    int idx = blockIdx.x * blockDim.x + threadIdx.x;
    if (idx >= BB * CL * N) return;
    int bc = idx / N;
    int r  = idx - bc * N;
    int y = r / WW, x = r - (r / WW) * WW;
    float fy = 4.0f * y + 1.5f;      // (y+0.5)*(256/64) - 0.5
    float fx = 4.0f * x + 1.5f;
    int y0 = (int)fy, x0 = (int)fx;
    float wy = fy - (float)y0, wx = fx - (float)x0;
    int y1 = y0 + 1; if (y1 > HL - 1) y1 = HL - 1;
    int x1 = x0 + 1; if (x1 > HL - 1) x1 = HL - 1;
    const float* base = low + (size_t)bc * HL * HL;
    float v00 = base[y0 * HL + x0], v01 = base[y0 * HL + x1];
    float v10 = base[y1 * HL + x0], v11 = base[y1 * HL + x1];
    lowr[idx] = (1.f - wy) * ((1.f - wx) * v00 + wx * v01)
              +        wy  * ((1.f - wx) * v10 + wx * v11);
}

// ---------------------------------------------------------------------------
// Nearest resize ROIs (B,256,256)->(B,64,64) as float + accumulate n_valid.
// ---------------------------------------------------------------------------
__global__ void resize_roi_kernel(const int* __restrict__ ROIs, float* __restrict__ rois,
                                  float* __restrict__ nvalid) {
    using namespace tel;
    int idx = blockIdx.x * blockDim.x + threadIdx.x;
    if (idx >= BB * N) return;
    int b = idx / N;
    int r = idx - b * N;
    int y = r / WW, x = r - (r / WW) * WW;
    float val = (float)ROIs[(size_t)b * HL * HL + (y * 4) * HL + (x * 4)];
    rois[idx] = val;
    atomicAdd(nvalid, val);
}

// ---------------------------------------------------------------------------
// Low-feature (3ch) edge weights: wt[e] = sum_c (emb[c,u]-emb[c,v])^2
// ---------------------------------------------------------------------------
__global__ void edge_w_low_kernel(const float* __restrict__ emb, float* __restrict__ wt) {
    using namespace tel;
    int e = blockIdx.x * blockDim.x + threadIdx.x;
    if (e >= E) return;
    int u, v; edge_uv(e, u, v);
    float acc = 0.f;
    #pragma unroll
    for (int c = 0; c < CL; ++c) {
        float d = emb[c * N + u] - emb[c * N + v];
        acc += d * d;
    }
    wt[e] = acc;
}

// ---------------------------------------------------------------------------
// High-feature (256ch) edge weights via WMMA reduction-by-ones.
// One wave per 16-edge tile. A[m][k] = f16((hi[k-th ch, u_m]-hi[...,v_m])^2),
// B = all-ones  =>  D[m][*] = row sums. Accumulate f32 over 8 chunks of 32 ch.
// Row m lives in lanes m and m+16 (two disjoint 16-channel halves); row sums
// are invariant to intra-row K permutation, so this mapping is exact.
// ---------------------------------------------------------------------------
__global__ __launch_bounds__(256)
void edge_w_high_wmma_kernel(const float* __restrict__ hi, float* __restrict__ wt) {
    using namespace tel;
    const int lane = threadIdx.x & 31;
    const int wave = threadIdx.x >> 5;
    const int tile = blockIdx.x * 8 + wave;     // 63 blocks * 8 waves = 504 tiles exactly
    const int row  = lane & 15;
    const int e    = tile * 16 + row;
    int u, v; edge_uv(e, u, v);
    const int chHalf = (lane < 16) ? 0 : 16;

    v16h ones;
    #pragma unroll
    for (int j = 0; j < 16; ++j) ones[j] = (_Float16)1.0f;

    v8f acc = {};
    #pragma unroll
    for (int k0 = 0; k0 < 8; ++k0) {
        // prefetch next chunk's row (CDNA5 global_prefetch_b8)
        __builtin_prefetch(&hi[(((k0 + 1) & 7) * 32 + chHalf) * N + u], 0, 1);
        v16h a;
        #pragma unroll
        for (int j = 0; j < 16; ++j) {
            int ch = k0 * 32 + chHalf + j;
            float d = hi[ch * N + u] - hi[ch * N + v];
            a[j] = (_Float16)(d * d);
        }
        acc = __builtin_amdgcn_wmma_f32_16x16x32_f16(
            /*neg_a=*/false, a, /*neg_b=*/false, ones,
            /*c_mod=*/(short)0, acc, /*reuse_a=*/false, /*reuse_b=*/false);
    }
    // D column 0: lane 0 holds rows 0..7 in acc[0..7]; lane 16 holds rows 8..15.
    if (lane == 0 || lane == 16) {
        int base = tile * 16 + ((lane == 16) ? 8 : 0);
        #pragma unroll
        for (int r = 0; r < 8; ++r) wt[base + r] = acc[r];
    }
}

// ---------------------------------------------------------------------------
// Boruvka MST, single workgroup. comp/best in LDS (48 KB), par/in_mst global.
// key = (weight_bits << 32) | edge_idx  — order-isomorphic to reference rank
// (weights are non-negative so IEEE bits are monotone).
// ---------------------------------------------------------------------------
__global__ __launch_bounds__(1024)
void boruvka_kernel(const float* __restrict__ wt, int* __restrict__ par,
                    int* __restrict__ in_mst) {
    using namespace tel;
    __shared__ int comp[N];
    __shared__ unsigned long long best[N];
    const int tid = threadIdx.x;

    for (int i = tid; i < N; i += 1024) comp[i] = i;
    for (int e = tid; e < E; e += 1024) in_mst[e] = 0;
    __syncthreads();

    for (int round = 0; round < 13; ++round) {
        for (int i = tid; i < N; i += 1024) { best[i] = ~0ull; par[i] = i; }
        __syncthreads();
        // scatter-min edge keys into both components
        for (int e = tid; e < E; e += 1024) {
            int u, v; edge_uv(e, u, v);
            int cu = comp[u], cv = comp[v];
            if (cu != cv) {
                unsigned long long key =
                    (((unsigned long long)__float_as_uint(wt[e])) << 32) | (unsigned)e;
                atomicMin(&best[cu], key);
                atomicMin(&best[cv], key);
            }
        }
        __syncthreads();
        // select winners, record merges (unique per component -> no race)
        for (int e = tid; e < E; e += 1024) {
            int u, v; edge_uv(e, u, v);
            int cu = comp[u], cv = comp[v];
            if (cu != cv) {
                unsigned long long key =
                    (((unsigned long long)__float_as_uint(wt[e])) << 32) | (unsigned)e;
                if (key == best[cu]) { in_mst[e] = 1; par[cu] = cv; }
                if (key == best[cv]) { in_mst[e] = 1; par[cv] = cu; }
            }
        }
        __syncthreads();
        // break 2-cycles: keep smaller index as root
        for (int i = tid; i < N; i += 1024) {
            int p = par[i];
            if (p != i && par[p] == i && i < p) par[i] = i;
        }
        __syncthreads();
        // pointer jumping
        for (int it = 0; it < 13; ++it) {
            int np[4]; int cnt = 0;
            for (int i = tid; i < N; i += 1024) np[cnt++] = par[par[i]];
            __syncthreads();
            cnt = 0;
            for (int i = tid; i < N; i += 1024) par[i] = np[cnt++];
            __syncthreads();
        }
        for (int i = tid; i < N; i += 1024) comp[i] = par[comp[i]];
        __syncthreads();
    }
}

// ---------------------------------------------------------------------------
// Root MST at node 0 via frontier BFS (double-buffered visited in LDS).
// ---------------------------------------------------------------------------
__global__ __launch_bounds__(1024)
void root_bfs_kernel(const int* __restrict__ in_mst, int* __restrict__ parent,
                     int* __restrict__ depth) {
    using namespace tel;
    __shared__ int vis[N];
    __shared__ int pend[N];
    __shared__ int changed;
    const int tid = threadIdx.x;
    for (int i = tid; i < N; i += 1024) {
        vis[i] = (i == 0) ? 1 : 0; pend[i] = 0; parent[i] = 0; depth[i] = 0;
    }
    __syncthreads();
    for (int iter = 0; iter < N + 2; ++iter) {
        if (tid == 0) changed = 0;
        __syncthreads();
        for (int e = tid; e < E; e += 1024) {
            if (!in_mst[e]) continue;
            int u, v; edge_uv(e, u, v);
            int vu = vis[u], vv = vis[v];
            if (vu && !vv)      { parent[v] = u; depth[v] = depth[u] + 1; pend[v] = 1; }
            else if (vv && !vu) { parent[u] = v; depth[u] = depth[v] + 1; pend[u] = 1; }
        }
        __syncthreads();
        for (int i = tid; i < N; i += 1024) {
            if (pend[i]) { vis[i] = 1; pend[i] = 0; changed = 1; }
        }
        __syncthreads();
        if (!changed) break;
    }
}

// ---------------------------------------------------------------------------
// Counting sort of nodes by depth -> level offsets + order array.
// ---------------------------------------------------------------------------
__global__ __launch_bounds__(1024)
void build_levels_kernel(const int* __restrict__ depth, int* __restrict__ cnt,
                         int* __restrict__ off, int* __restrict__ order,
                         int* __restrict__ nlv) {
    using namespace tel;
    const int tid = threadIdx.x;
    for (int i = tid; i <= N; i += 1024) cnt[i] = 0;
    __syncthreads();
    for (int i = tid; i < N; i += 1024) atomicAdd(&cnt[depth[i]], 1);
    __syncthreads();
    if (tid == 0) {
        int acc = 0, mx = 0;
        for (int d = 0; d <= N; ++d) {
            off[d] = acc; acc += cnt[d];
            if (cnt[d] > 0) mx = d;
        }
        off[N + 1] = acc;
        *nlv = mx + 1;
    }
    __syncthreads();
    for (int i = tid; i <= N; i += 1024) cnt[i] = off[i];   // reuse as scatter ptr
    __syncthreads();
    for (int i = tid; i < N; i += 1024) {
        int pos = atomicAdd(&cnt[depth[i]], 1);
        order[pos] = i;
    }
}

// ---------------------------------------------------------------------------
// Per-node parent-edge affinity: w[i] = exp(-||emb_i - emb_parent||^2 / sigma)
// ---------------------------------------------------------------------------
__global__ void wnode_kernel(const float* __restrict__ emb, int Cemb, float sigma,
                             const int* __restrict__ parent, float* __restrict__ wnode) {
    using namespace tel;
    int i = blockIdx.x * blockDim.x + threadIdx.x;
    if (i >= N) return;
    int p = parent[i];
    float d2 = 0.f;
    for (int ch = 0; ch < Cemb; ++ch) {
        float d = emb[ch * N + i] - emb[ch * N + p];
        d2 += d * d;
    }
    wnode[i] = (p == i) ? 0.f : expf(-d2 / sigma);
}

// ---------------------------------------------------------------------------
// S_up init: [feat(21ch); ones] per node, node-major [N][22].
// ---------------------------------------------------------------------------
__global__ void init_S_kernel(const float* __restrict__ feat, float* __restrict__ Sup) {
    using namespace tel;
    int i = blockIdx.x * blockDim.x + threadIdx.x;
    if (i >= N) return;
    float* s = Sup + i * (C + 1);
    #pragma unroll
    for (int c = 0; c < C; ++c) s[c] = feat[c * N + i];
    s[C] = 1.0f;
}

// Leaf->root: atomicAdd children into parents, level by level (deep -> shallow).
__global__ __launch_bounds__(1024)
void up_sweep_kernel(float* __restrict__ Sup, const float* __restrict__ wnode,
                     const int* __restrict__ parent, const int* __restrict__ order,
                     const int* __restrict__ off, const int* __restrict__ nlv) {
    using namespace tel;
    const int tid = threadIdx.x;
    const int L = *nlv;
    for (int lvl = L - 1; lvl >= 1; --lvl) {
        int s0 = off[lvl], s1 = off[lvl + 1];
        for (int idx = s0 + tid; idx < s1; idx += 1024) {
            int i = order[idx]; int p = parent[i]; float w = wnode[i];
            float* Si = Sup + i * (C + 1);
            float* Sp = Sup + p * (C + 1);
            #pragma unroll
            for (int c = 0; c <= C; ++c) atomicAdd(&Sp[c], w * Si[c]);
        }
        __syncthreads();
    }
}

// Root->leaf: S[i] = Sup[i] + w*(S[parent] - w*Sup[i]), level by level.
__global__ __launch_bounds__(1024)
void down_sweep_kernel(float* __restrict__ S, const float* __restrict__ Sup,
                       const float* __restrict__ wnode, const int* __restrict__ parent,
                       const int* __restrict__ order, const int* __restrict__ off,
                       const int* __restrict__ nlv) {
    using namespace tel;
    const int tid = threadIdx.x;
    for (int t = tid; t < N * (C + 1); t += 1024) S[t] = Sup[t];
    __syncthreads();
    const int L = *nlv;
    for (int lvl = 1; lvl < L; ++lvl) {
        int s0 = off[lvl], s1 = off[lvl + 1];
        for (int idx = s0 + tid; idx < s1; idx += 1024) {
            int i = order[idx]; int p = parent[i]; float w = wnode[i];
            const float* Ui = Sup + i * (C + 1);
            const float* Sp = S + p * (C + 1);
            float* Si = S + i * (C + 1);
            #pragma unroll
            for (int c = 0; c <= C; ++c) Si[c] = Ui[c] + w * (Sp[c] - w * Ui[c]);
        }
        __syncthreads();
    }
}

// Normalize: AS[c,i] = S[i][c] / S[i][21]
__global__ void normalize_kernel(const float* __restrict__ S, float* __restrict__ AS) {
    using namespace tel;
    int idx = blockIdx.x * blockDim.x + threadIdx.x;
    if (idx >= C * N) return;
    int c = idx / N;
    int i = idx - c * N;
    AS[idx] = S[i * (C + 1) + c] / S[i * (C + 1) + C];
}

// Loss: sum rois[i] * |preds[c,i] - AS2[c,i]|  (block-reduced atomic add)
__global__ __launch_bounds__(256)
void loss_kernel(const float* __restrict__ preds_b, const float* __restrict__ rois_b,
                 const float* __restrict__ AS2, float* __restrict__ lossAcc) {
    using namespace tel;
    __shared__ float red[256];
    int idx = blockIdx.x * 256 + threadIdx.x;
    float v = 0.f;
    if (idx < C * N) {
        int i = idx % N;
        v = rois_b[i] * fabsf(preds_b[idx] - AS2[idx]);
    }
    red[threadIdx.x] = v;
    __syncthreads();
    for (int s = 128; s > 0; s >>= 1) {
        if (threadIdx.x < s) red[threadIdx.x] += red[threadIdx.x + s];
        __syncthreads();
    }
    if (threadIdx.x == 0) atomicAdd(lossAcc, red[0]);
}

__global__ void finalize_kernel(const float* __restrict__ lossAcc,
                                const float* __restrict__ nvalid,
                                const int* __restrict__ tw, float* __restrict__ out) {
    __builtin_amdgcn_s_wait_tensorcnt(0);   // CDNA5 split-counter wait (no-op here)
    if (threadIdx.x == 0 && blockIdx.x == 0) {
        float l = *lossAcc, nv = *nvalid;
        float res = (nv > 0.f) ? l / nv : l;
        out[0] = (float)(*tw) * res;
    }
}

// ---------------------------------------------------------------------------
// Host orchestration
// ---------------------------------------------------------------------------
extern "C" void kernel_launch(void* const* d_in, const int* in_sizes, int n_in,
                              void* d_out, int out_size, void* d_ws, size_t ws_size,
                              hipStream_t stream) {
    using namespace tel;
    (void)in_sizes; (void)n_in; (void)out_size; (void)ws_size;

    const float* preds = (const float*)d_in[0];   // B,21,64,64
    const float* lowf  = (const float*)d_in[1];   // B,3,256,256
    const float* highf = (const float*)d_in[2];   // B,256,64,64
    const int*   ROIs  = (const int*)d_in[3];     // B,256,256
    const int*   tw    = (const int*)d_in[4];     // scalar
    float* out = (float*)d_out;

    // Workspace layout (~1.75 MB total)
    float* f      = (float*)d_ws;
    float* lowr   = f;                       // B*3*N
    float* rois   = lowr  + BB * CL * N;     // B*N
    float* nvalid = rois  + BB * N;          // 1
    float* lossA  = nvalid + 1;              // 1
    float* wt     = lossA + 1;               // E
    float* wnode  = wt    + E;               // N
    float* S      = wnode + N;               // (C+1)*N
    float* Sup    = S     + (C + 1) * N;     // (C+1)*N
    float* AS1    = Sup   + (C + 1) * N;     // C*N
    float* AS2    = AS1   + C * N;           // C*N
    int*  ipar    = (int*)(AS2 + C * N);     // N+1
    int*  inmst   = ipar   + (N + 1);        // E
    int*  parent  = inmst  + E;              // N
    int*  depth   = parent + N;              // N
    int*  lvlcnt  = depth  + N;              // N+1
    int*  lvloff  = lvlcnt + (N + 1);        // N+2
    int*  order   = lvloff + (N + 2);        // N
    int*  nlv     = order  + N;              // 1

    zero2_kernel<<<1, 1, 0, stream>>>(lossA, nvalid);
    resize_low_kernel<<<(BB * CL * N + 255) / 256, 256, 0, stream>>>(lowf, lowr);
    resize_roi_kernel<<<(BB * N + 255) / 256, 256, 0, stream>>>(ROIs, rois, nvalid);

    for (int b = 0; b < BB; ++b) {
        const float* lowr_b  = lowr  + (size_t)b * CL * N;
        const float* high_b  = highf + (size_t)b * CH * N;
        const float* preds_b = preds + (size_t)b * C * N;
        const float* rois_b  = rois  + (size_t)b * N;

        // ---- low tree: build + filter preds -> AS1 (sigma = 0.002)
        edge_w_low_kernel<<<(E + 255) / 256, 256, 0, stream>>>(lowr_b, wt);
        boruvka_kernel<<<1, 1024, 0, stream>>>(wt, ipar, inmst);
        root_bfs_kernel<<<1, 1024, 0, stream>>>(inmst, parent, depth);
        build_levels_kernel<<<1, 1024, 0, stream>>>(depth, lvlcnt, lvloff, order, nlv);
        wnode_kernel<<<(N + 255) / 256, 256, 0, stream>>>(lowr_b, CL, SIGMA_LOW, parent, wnode);
        init_S_kernel<<<(N + 255) / 256, 256, 0, stream>>>(preds_b, Sup);
        up_sweep_kernel<<<1, 1024, 0, stream>>>(Sup, wnode, parent, order, lvloff, nlv);
        down_sweep_kernel<<<1, 1024, 0, stream>>>(S, Sup, wnode, parent, order, lvloff, nlv);
        normalize_kernel<<<(C * N + 255) / 256, 256, 0, stream>>>(S, AS1);

        // ---- high tree: build (WMMA edge weights) + filter AS1 -> AS2 (sigma = 1)
        edge_w_high_wmma_kernel<<<63, 256, 0, stream>>>(high_b, wt);
        boruvka_kernel<<<1, 1024, 0, stream>>>(wt, ipar, inmst);
        root_bfs_kernel<<<1, 1024, 0, stream>>>(inmst, parent, depth);
        build_levels_kernel<<<1, 1024, 0, stream>>>(depth, lvlcnt, lvloff, order, nlv);
        wnode_kernel<<<(N + 255) / 256, 256, 0, stream>>>(high_b, CH, SIGMA_HIGH, parent, wnode);
        init_S_kernel<<<(N + 255) / 256, 256, 0, stream>>>(AS1, Sup);
        up_sweep_kernel<<<1, 1024, 0, stream>>>(Sup, wnode, parent, order, lvloff, nlv);
        down_sweep_kernel<<<1, 1024, 0, stream>>>(S, Sup, wnode, parent, order, lvloff, nlv);
        normalize_kernel<<<(C * N + 255) / 256, 256, 0, stream>>>(S, AS2);

        loss_kernel<<<(C * N + 255) / 256, 256, 0, stream>>>(preds_b, rois_b, AS2, lossA);
    }

    finalize_kernel<<<1, 1, 0, stream>>>(lossA, nvalid, tw, out);
}
